// TritonFlashCausalAttention_46918222741518
// MI455X (gfx1250) — matrix operand order repaired and checked by reading, not compiled, not run on hardware
//
#include <hip/hip_runtime.h>
#include <hip/hip_bf16.h>
#include <math.h>

#define D_MODEL 2048
#define N_HEADS 16
#define BATCH 2
#define SEQ 2048
#define HEAD_DIM 128
#define BS (BATCH * SEQ)   /* 4096 rows */
#define D3 (3 * D_MODEL)   /* 6144 */

typedef __bf16 bf16_t;
typedef __attribute__((ext_vector_type(16))) __bf16 v16bf;
typedef __attribute__((ext_vector_type(8)))  __bf16 v8bf;
typedef __attribute__((ext_vector_type(8)))  float  v8f;

union ABFrag { v16bf v; v8bf h[2]; };

static __device__ __forceinline__ v8f wmma_bf16(v16bf a, v16bf b, v8f c) {
  // D = A(16x32 bf16) * B(32x16 bf16) + C(16x16 f32)
  return __builtin_amdgcn_wmma_f32_16x16x32_bf16(
      /*neg_a=*/false, a, /*neg_b=*/false, b,
      /*c_mod=*/(short)0, c, /*reuse_a=*/false, /*reuse_b=*/false);
}

static __device__ __forceinline__ unsigned short f2bfbits(float f) {
  unsigned u = __float_as_uint(f);
  return (unsigned short)((u + 0x7FFFu + ((u >> 16) & 1u)) >> 16); // RNE
}
static __device__ __forceinline__ bf16_t f2bf(float f) {
  unsigned short s = f2bfbits(f);
  bf16_t b;
  __builtin_memcpy(&b, &s, 2);
  return b;
}

// ---------------------------------------------------------------------------
// fp32 -> bf16 cast (vectorized 4-wide, grid-stride)
// ---------------------------------------------------------------------------
__global__ void cvt_f32_to_bf16(const float* __restrict__ in,
                                bf16_t* __restrict__ out, int n4) {
  int i = blockIdx.x * blockDim.x + threadIdx.x;
  int stride = gridDim.x * blockDim.x;
  for (; i < n4; i += stride) {
    float4 f = ((const float4*)in)[i];
    ushort4 u;
    u.x = f2bfbits(f.x); u.y = f2bfbits(f.y);
    u.z = f2bfbits(f.z); u.w = f2bfbits(f.w);
    ((ushort4*)out)[i] = u;
  }
}

// ---------------------------------------------------------------------------
// QKV GEMM: qkv[m, n] = sum_k x[m,k] * w_qkv[n,k] + b_qkv[n]
// 1 wave -> 16(M) x 64(N) strip, K stepped by 32, register double-buffered.
// Epilogue scatters into q:[B,H,S,hd]  k:[B,H,S,hd]  vT:[B,H,hd,S] (bf16).
// ---------------------------------------------------------------------------
__global__ void __launch_bounds__(128)
qkv_gemm(const bf16_t* __restrict__ xb,      // [BS, D]
         const bf16_t* __restrict__ wb,      // [3D, D]
         const float*  __restrict__ bqkv,    // [3D]
         bf16_t* __restrict__ qbuf,
         bf16_t* __restrict__ kbuf,
         bf16_t* __restrict__ vTbuf) {
  const int wave = threadIdx.x >> 5;
  const int lane = threadIdx.x & 31;
  const int lr = lane & 15;
  const int lh = lane >> 4;

  const int m0 = blockIdx.y * 16;
  const int n0 = (blockIdx.x * 4 + wave) * 64;

  v8f acc[4];
#pragma unroll
  for (int j = 0; j < 4; ++j) acc[j] = (v8f){};

  const bf16_t* arow = xb + (size_t)(m0 + lr) * D_MODEL;
  const bf16_t* brow0 = wb + (size_t)(n0 + lr) * D_MODEL + lh * 16;

  // Prologue loads for k0 = 0.
  ABFrag a0;
  a0.h[0] = *(const v8bf*)(arow + lh * 16);
  a0.h[1] = *(const v8bf*)(arow + lh * 16 + 8);
  v16bf b0[4];
#pragma unroll
  for (int j = 0; j < 4; ++j)
    b0[j] = *(const v16bf*)(brow0 + (size_t)(j * 16) * D_MODEL);

#pragma unroll 2
  for (int k0 = 0; k0 < D_MODEL; k0 += 32) {
    const int k1 = (k0 + 32) & (D_MODEL - 1);  // wraps to 0 on last iter
    __builtin_prefetch(arow + k0 + 256, 0, 3);
    // Issue next-step loads before consuming current fragments.
    ABFrag a1;
    a1.h[0] = *(const v8bf*)(arow + k1 + lh * 16);
    a1.h[1] = *(const v8bf*)(arow + k1 + lh * 16 + 8);
    v16bf b1[4];
#pragma unroll
    for (int j = 0; j < 4; ++j)
      b1[j] = *(const v16bf*)(brow0 + (size_t)(j * 16) * D_MODEL + k1);
#pragma unroll
    for (int j = 0; j < 4; ++j)
      acc[j] = wmma_bf16(a0.v, b0[j], acc[j]);
    a0 = a1;
#pragma unroll
    for (int j = 0; j < 4; ++j) b0[j] = b1[j];
  }

  // Epilogue: add bias, convert, scatter into per-head layouts.
#pragma unroll
  for (int j = 0; j < 4; ++j) {
#pragma unroll
    for (int r = 0; r < 8; ++r) {
      const int m = m0 + r + lh * 8;          // row in [0, BS)
      const int c = n0 + j * 16 + lr;         // col in [0, 3D)
      const float val = acc[j][r] + bqkv[c];
      const bf16_t bv = f2bf(val);
      const int part = c >> 11;               // 0=q 1=k 2=v
      const int within = c & (D_MODEL - 1);
      const int h = within >> 7;
      const int d = within & (HEAD_DIM - 1);
      const int b = m >> 11;                  // m / SEQ
      const int s = m & (SEQ - 1);
      const size_t bh = (size_t)b * N_HEADS + h;
      if (part == 0) {
        qbuf[(bh * SEQ + s) * HEAD_DIM + d] = bv;
      } else if (part == 1) {
        kbuf[(bh * SEQ + s) * HEAD_DIM + d] = bv;
      } else {
        vTbuf[(bh * HEAD_DIM + d) * SEQ + s] = bv;  // transposed for PV B-frags
      }
    }
  }
}

// ---------------------------------------------------------------------------
// Flash attention with causal mask + ALiBi. One wave per 16-row Q tile.
// All K and V fragments for a 32-key block are issued in one clause up front;
// score WMMAs hide K latency, softmax VALU hides V latency.
// ---------------------------------------------------------------------------
__global__ void __launch_bounds__(128)
flash_attn(const bf16_t* __restrict__ qb,
           const bf16_t* __restrict__ kb,
           const bf16_t* __restrict__ vT,
           bf16_t* __restrict__ ctxb) {
  __shared__ __attribute__((aligned(64))) bf16_t ptile_all[4][16 * 32];

  const int wave = threadIdx.x >> 5;
  const int lane = threadIdx.x & 31;
  const int lr = lane & 15;
  const int lh = lane >> 4;

  const int bh = blockIdx.x;              // [0, B*H)
  const int h = bh & (N_HEADS - 1);
  const int qt = blockIdx.y * 4 + wave;   // q tile [0, 128)
  const int q0 = qt * 16;

  const bf16_t* qptr = qb + (size_t)bh * SEQ * HEAD_DIM;
  const bf16_t* kptr = kb + (size_t)bh * SEQ * HEAD_DIM;
  const bf16_t* vptr = vT + (size_t)bh * HEAD_DIM * SEQ;
  bf16_t* pt = ptile_all[wave];

  // Load Q A-fragments for the whole hd=128 (4 K-steps of 32).
  ABFrag qa[4];
  const int rowQ = q0 + lr;
#pragma unroll
  for (int t = 0; t < 4; ++t) {
    qa[t].h[0] = *(const v8bf*)(qptr + (size_t)rowQ * HEAD_DIM + t * 32 + lh * 16);
    qa[t].h[1] = *(const v8bf*)(qptr + (size_t)rowQ * HEAD_DIM + t * 32 + lh * 16 + 8);
  }

  v8f ctx[8];
#pragma unroll
  for (int f = 0; f < 8; ++f) ctx[f] = (v8f){};

  float mrow[8], lrow[8];
#pragma unroll
  for (int r = 0; r < 8; ++r) { mrow[r] = -INFINITY; lrow[r] = 0.0f; }

  const float slope = exp2f(-(float)(h + 1) * (8.0f / (float)N_HEADS));
  const float scale = 0.088388347648318447f;  // 1/sqrt(128)

  const int nblocks = (q0 + 16 + 31) >> 5;  // 32-key blocks covering [0, q0+16)
  for (int blk = 0; blk < nblocks; ++blk) {
    const int kbase = blk * 32;

    // Issue ALL loads for this block up front (K frags then V frags).
    v16bf kf[8];
#pragma unroll
    for (int t = 0; t < 4; ++t) {
      kf[2 * t] =
          *(const v16bf*)(kptr + (size_t)(kbase + lr) * HEAD_DIM + t * 32 + lh * 16);
      kf[2 * t + 1] =
          *(const v16bf*)(kptr + (size_t)(kbase + 16 + lr) * HEAD_DIM + t * 32 + lh * 16);
    }
    v16bf vf[8];
#pragma unroll
    for (int f = 0; f < 8; ++f)
      vf[f] = *(const v16bf*)(vptr + (size_t)(f * 16 + lr) * SEQ + kbase + lh * 16);
    __builtin_prefetch(kptr + (size_t)(kbase + 32 + lr) * HEAD_DIM, 0, 3);

    // Scores: 16 q-rows x 32 keys as two 16-wide C fragments.
    v8f s0 = (v8f){}, s1 = (v8f){};
#pragma unroll
    for (int t = 0; t < 4; ++t) {
      s0 = wmma_bf16(qa[t].v, kf[2 * t], s0);
      s1 = wmma_bf16(qa[t].v, kf[2 * t + 1], s1);
    }

    // Online softmax. C layout: lane<16 holds row (q0+r), lane>=16 row (q0+r+8),
    // column = kbase + (lane&15) (+16 for s1) -> per-lane stats line up per row.
    const int kcol0 = kbase + lr;
    const int kcol1 = kcol0 + 16;
    const float bias0 = slope * (float)(kcol0 - (SEQ - 1));
    const float bias1 = slope * (float)(kcol1 - (SEQ - 1));
    float e0s[8], e1s[8], alphar[8];
#pragma unroll
    for (int r = 0; r < 8; ++r) {
      const int qrow = q0 + r + lh * 8;
      float a0 = (kcol0 <= qrow) ? s0[r] * scale + bias0 : -INFINITY;
      float a1 = (kcol1 <= qrow) ? s1[r] * scale + bias1 : -INFINITY;
      float mx = fmaxf(a0, a1);
#pragma unroll
      for (int off = 1; off < 16; off <<= 1) mx = fmaxf(mx, __shfl_xor(mx, off, 32));
      const float mnew = fmaxf(mrow[r], mx);
      float alpha = __expf(mrow[r] - mnew);
      if (!(mnew > -INFINITY)) alpha = 1.0f;  // fully-masked guard
      float e0 = (a0 > -INFINITY) ? __expf(a0 - mnew) : 0.0f;
      float e1 = (a1 > -INFINITY) ? __expf(a1 - mnew) : 0.0f;
      float sum = e0 + e1;
#pragma unroll
      for (int off = 1; off < 16; off <<= 1) sum += __shfl_xor(sum, off, 32);
      lrow[r] = lrow[r] * alpha + sum;
      mrow[r] = mnew;
      e0s[r] = e0; e1s[r] = e1; alphar[r] = alpha;
    }

    // Rescale running context by alpha (per-row, matches C layout).
#pragma unroll
    for (int f = 0; f < 8; ++f)
#pragma unroll
      for (int r = 0; r < 8; ++r) ctx[f][r] *= alphar[r];

    // C-layout P -> LDS (row-major 16x32 bf16) -> A-layout fragment.
#pragma unroll
    for (int r = 0; r < 8; ++r) {
      const int row = r + lh * 8;
      pt[row * 32 + lr] = f2bf(e0s[r]);
      pt[row * 32 + 16 + lr] = f2bf(e1s[r]);
    }
    asm volatile("s_wait_dscnt 0x0" ::: "memory");
    ABFrag pa;
    pa.h[0] = *(const v8bf*)(pt + lr * 32 + lh * 16);
    pa.h[1] = *(const v8bf*)(pt + lr * 32 + lh * 16 + 8);

    // PV: accumulate ctx += P(16x32) * V(32x128), 8 WMMAs over dim groups.
#pragma unroll
    for (int f = 0; f < 8; ++f)
      ctx[f] = wmma_bf16(pa.v, vf[f], ctx[f]);
  }

  // Normalize and store ctx (bf16, layout [B, S, H*hd] == [BS, D]).
  const int batch = bh >> 4;  // bh / N_HEADS
#pragma unroll
  for (int f = 0; f < 8; ++f) {
#pragma unroll
    for (int r = 0; r < 8; ++r) {
      const int qrow = q0 + r + lh * 8;
      const float val = ctx[f][r] * (1.0f / lrow[r]);
      const int d = f * 16 + lr;
      ctxb[(((size_t)batch * SEQ + qrow) * N_HEADS + h) * HEAD_DIM + d] = f2bf(val);
    }
  }
}

// ---------------------------------------------------------------------------
// Output GEMM: out[m,n] = sum_k ctx[m,k] * w_out[n,k] + b_out[n]  (f32 out)
// Same 16x64-per-wave, register double-buffered structure.
// ---------------------------------------------------------------------------
__global__ void __launch_bounds__(128)
out_gemm(const bf16_t* __restrict__ ctxb,   // [BS, D]
         const bf16_t* __restrict__ wob,    // [D, D]
         const float* __restrict__ bout,    // [D]
         float* __restrict__ out) {         // [BS, D]
  const int wave = threadIdx.x >> 5;
  const int lane = threadIdx.x & 31;
  const int lr = lane & 15;
  const int lh = lane >> 4;

  const int m0 = blockIdx.y * 16;
  const int n0 = (blockIdx.x * 4 + wave) * 64;

  v8f acc[4];
#pragma unroll
  for (int j = 0; j < 4; ++j) acc[j] = (v8f){};

  const bf16_t* arow = ctxb + (size_t)(m0 + lr) * D_MODEL;
  const bf16_t* brow0 = wob + (size_t)(n0 + lr) * D_MODEL + lh * 16;

  ABFrag a0;
  a0.h[0] = *(const v8bf*)(arow + lh * 16);
  a0.h[1] = *(const v8bf*)(arow + lh * 16 + 8);
  v16bf b0[4];
#pragma unroll
  for (int j = 0; j < 4; ++j)
    b0[j] = *(const v16bf*)(brow0 + (size_t)(j * 16) * D_MODEL);

#pragma unroll 2
  for (int k0 = 0; k0 < D_MODEL; k0 += 32) {
    const int k1 = (k0 + 32) & (D_MODEL - 1);
    __builtin_prefetch(arow + k0 + 256, 0, 3);
    ABFrag a1;
    a1.h[0] = *(const v8bf*)(arow + k1 + lh * 16);
    a1.h[1] = *(const v8bf*)(arow + k1 + lh * 16 + 8);
    v16bf b1[4];
#pragma unroll
    for (int j = 0; j < 4; ++j)
      b1[j] = *(const v16bf*)(brow0 + (size_t)(j * 16) * D_MODEL + k1);
#pragma unroll
    for (int j = 0; j < 4; ++j)
      acc[j] = wmma_bf16(a0.v, b0[j], acc[j]);
    a0 = a1;
#pragma unroll
    for (int j = 0; j < 4; ++j) b0[j] = b1[j];
  }

#pragma unroll
  for (int j = 0; j < 4; ++j) {
#pragma unroll
    for (int r = 0; r < 8; ++r) {
      const int m = m0 + r + lh * 8;
      const int c = n0 + j * 16 + lr;
      out[(size_t)m * D_MODEL + c] = acc[j][r] + bout[c];
    }
  }
}

// ---------------------------------------------------------------------------
extern "C" void kernel_launch(void* const* d_in, const int* in_sizes, int n_in,
                              void* d_out, int out_size, void* d_ws, size_t ws_size,
                              hipStream_t stream) {
  (void)in_sizes; (void)n_in; (void)out_size; (void)ws_size;
  const float* x     = (const float*)d_in[0];
  // d_in[1] = attn_mask (ALiBi) — recomputed analytically on device
  const float* w_qkv = (const float*)d_in[2];
  const float* b_qkv = (const float*)d_in[3];
  const float* w_out = (const float*)d_in[4];
  const float* b_out = (const float*)d_in[5];
  float* out = (float*)d_out;

  char* ws = (char*)d_ws;
  size_t off = 0;
  auto alloc = [&](size_t bytes) -> void* {
    void* p = ws + off;
    off += (bytes + 255) & ~(size_t)255;
    return p;
  };

  bf16_t* xb    = (bf16_t*)alloc((size_t)BS * D_MODEL * sizeof(bf16_t));
  bf16_t* wqkvb = (bf16_t*)alloc((size_t)D3 * D_MODEL * sizeof(bf16_t));
  bf16_t* woutb = (bf16_t*)alloc((size_t)D_MODEL * D_MODEL * sizeof(bf16_t));
  bf16_t* qbuf  = (bf16_t*)alloc((size_t)BATCH * N_HEADS * SEQ * HEAD_DIM * sizeof(bf16_t));
  bf16_t* kbuf  = (bf16_t*)alloc((size_t)BATCH * N_HEADS * SEQ * HEAD_DIM * sizeof(bf16_t));
  bf16_t* vTbuf = (bf16_t*)alloc((size_t)BATCH * N_HEADS * HEAD_DIM * SEQ * sizeof(bf16_t));
  bf16_t* ctxb  = (bf16_t*)alloc((size_t)BS * D_MODEL * sizeof(bf16_t));

  cvt_f32_to_bf16<<<1024, 256, 0, stream>>>(x, xb, BS * D_MODEL / 4);
  cvt_f32_to_bf16<<<1024, 256, 0, stream>>>(w_qkv, wqkvb, D3 * D_MODEL / 4);
  cvt_f32_to_bf16<<<512, 256, 0, stream>>>(w_out, woutb, D_MODEL * D_MODEL / 4);

  // QKV: N strips of 64 over 3D=6144 -> 96 strips, 4 waves/block -> 24 blocks x;
  //      M tiles of 16 over BS=4096 -> 256 blocks y.
  qkv_gemm<<<dim3(24, 256), 128, 0, stream>>>(xb, wqkvb, b_qkv, qbuf, kbuf, vTbuf);

  // Attention: x = B*H = 32 ; y = (S/16)/4 = 32 (4 q-tiles per block).
  flash_attn<<<dim3(32, 32), 128, 0, stream>>>(qbuf, kbuf, vTbuf, ctxb);

  // Output GEMM: N strips of 64 over 2048 -> 32 strips / 4 waves -> 8 x; 256 y.
  out_gemm<<<dim3(8, 256), 128, 0, stream>>>(ctxb, woutb, b_out, out);
}
